// LlamaAttention_21655225107095
// MI455X (gfx1250) — matrix operand order repaired and checked by reading, not compiled, not run on hardware
//
#include <hip/hip_runtime.h>
#include <stdint.h>

// ---------------------------------------------------------------------------
// Types for WMMA fragments (CDNA5 / gfx1250, wave32)
// ---------------------------------------------------------------------------
typedef __attribute__((ext_vector_type(16))) __bf16   v16bf;
typedef __attribute__((ext_vector_type(8)))  float    v8f;
typedef __attribute__((ext_vector_type(4)))  unsigned int u32x4;

union BF16Frag {
    v16bf v;
    u32x4 q[2];
    unsigned short h[16];
};

__device__ __forceinline__ unsigned short f2bf(float f) {
    union { float f; unsigned int u; } c; c.f = f;
    unsigned int u = c.u + 0x7fffu + ((c.u >> 16) & 1u);   // round-to-nearest-even
    return (unsigned short)(u >> 16);
}

__device__ __forceinline__ v8f zero8() {
    v8f z;
    #pragma unroll
    for (int i = 0; i < 8; ++i) z[i] = 0.0f;
    return z;
}

// CDNA5 async global->LDS copy (ASYNCcnt-tracked, no VGPR round-trip)
__device__ __forceinline__ void async_copy16(unsigned int lds_addr, const void* gaddr) {
    asm volatile("global_load_async_to_lds_b128 %0, %1, off"
                 :: "v"(lds_addr), "v"(gaddr) : "memory");
}
__device__ __forceinline__ void wait_async0() {
    asm volatile("s_wait_asynccnt 0x0" ::: "memory");
}

#define SEQ     2048
#define HID     2048
#define NHEADS  32
#define NKV     8
#define HDIM    64
#define KVDIM   (NKV * HDIM)   // 512

// GEMM block tiling
#define BM   64
#define BN   128
#define BK   64
#define BKP  72    // padded LDS row (halves) to break bank conflicts

// ---------------------------------------------------------------------------
// f32 -> bf16 conversion
// ---------------------------------------------------------------------------
__global__ void llama_cvt_bf16(const float* __restrict__ src,
                               unsigned short* __restrict__ dst, int n) {
    int i = blockIdx.x * blockDim.x + threadIdx.x;
    if (i < n) dst[i] = f2bf(src[i]);
}

// ---------------------------------------------------------------------------
// LDS-staged, double-buffered GEMM with async tile copies:
//   C[MxN] (f32, row-major) = A[MxK] (bf16) @ B[NxK]^T (bf16, rows over K)
// 4 waves / block; block tile 64x128; each wave computes 32x64 (8 WMMA tiles).
// ---------------------------------------------------------------------------
__device__ __forceinline__ void async_tile(const unsigned short* __restrict__ A,
                                           const unsigned short* __restrict__ B,
                                           unsigned short* As, unsigned short* Bs,
                                           int m0, int n0, int K, int kc, int t) {
    #pragma unroll
    for (int i = 0; i < 4; ++i) {           // A: 64 rows x 64 halves = 512 x 16B chunks
        int c = i * 128 + t;
        int row = c >> 3, col = (c & 7) * 8;
        async_copy16((unsigned int)(uintptr_t)(As + row * BKP + col),
                     A + (size_t)(m0 + row) * K + kc + col);
    }
    #pragma unroll
    for (int i = 0; i < 8; ++i) {           // B: 128 rows x 64 halves = 1024 x 16B chunks
        int c = i * 128 + t;
        int row = c >> 3, col = (c & 7) * 8;
        async_copy16((unsigned int)(uintptr_t)(Bs + row * BKP + col),
                     B + (size_t)(n0 + row) * K + kc + col);
    }
}

// Load ALL fragments of the BK=64 chunk first (24 x ds_load_b128), then run
// the 16 WMMAs as a dense block: one partial-wait cascade instead of an
// s_wait_dscnt 0x0 stall before every WMMA pair.
__device__ __forceinline__ void compute64(const unsigned short* As,
                                          const unsigned short* Bs,
                                          int msub, int nsub, int lg, int lr,
                                          v8f acc[2][4]) {
    v16bf af[2][2];   // [k-step][m-tile]
    v16bf bf[2][4];   // [k-step][n-tile]

    #pragma unroll
    for (int ks2 = 0; ks2 < 2; ++ks2) {
        const int ks = ks2 * 32;
        #pragma unroll
        for (int mt = 0; mt < 2; ++mt) {
            // A 16x32: lane<16 -> M=lane, K {0..15}; lane>=16 -> K {16..31}
            BF16Frag a;
            const unsigned short* p = As + (msub + mt * 16 + lr) * BKP + ks + lg * 16;
            a.q[0] = *(const u32x4*)p;
            a.q[1] = *(const u32x4*)(p + 8);
            af[ks2][mt] = a.v;
        }
        #pragma unroll
        for (int nt = 0; nt < 4; ++nt) {
            // B 32x16: lane<16 -> N=lane, K {0..15}; lane>=16 -> K {16..31}
            BF16Frag b;
            const unsigned short* p = Bs + (nsub + nt * 16 + lr) * BKP + ks + lg * 16;
            b.q[0] = ((const u32x4*)p)[0];
            b.q[1] = ((const u32x4*)p)[1];
            bf[ks2][nt] = b.v;
        }
    }

    #pragma unroll
    for (int ks2 = 0; ks2 < 2; ++ks2)
        #pragma unroll
        for (int nt = 0; nt < 4; ++nt)
            #pragma unroll
            for (int mt = 0; mt < 2; ++mt)
                acc[mt][nt] = __builtin_amdgcn_wmma_f32_16x16x32_bf16(
                    false, af[ks2][mt], false, bf[ks2][nt], (short)0,
                    acc[mt][nt], false, false);
}

__global__ __launch_bounds__(128)
void llama_gemm_bf16(const unsigned short* __restrict__ A,
                     const unsigned short* __restrict__ B,
                     float* __restrict__ C, int M, int N, int K) {
    __shared__ __attribute__((aligned(16))) unsigned short As[2][BM * BKP];
    __shared__ __attribute__((aligned(16))) unsigned short Bs[2][BN * BKP];

    const int t    = threadIdx.x;
    const int n0   = blockIdx.x * BN;
    const int m0   = blockIdx.y * BM;
    const int w    = t >> 5;
    const int l    = t & 31;
    const int lg   = l >> 4, lr = l & 15;
    const int msub = (w >> 1) * 32;      // wave's 32-row slice of the 64-row block
    const int nsub = (w & 1) * 64;       // wave's 64-col slice of the 128-col block

    v8f acc[2][4];
    #pragma unroll
    for (int mt = 0; mt < 2; ++mt)
        #pragma unroll
        for (int nt = 0; nt < 4; ++nt) acc[mt][nt] = zero8();

    async_tile(A, B, As[0], Bs[0], m0, n0, K, 0, t);
    wait_async0();
    __syncthreads();

    int buf = 0;
    for (int kc = BK; kc < K; kc += BK) {
        async_tile(A, B, As[buf ^ 1], Bs[buf ^ 1], m0, n0, K, kc, t);  // in flight
        if (kc + BK < K) {                          // prefetch chunk after next into L2
            __builtin_prefetch(A + (size_t)(m0 + (t >> 1)) * K + kc + BK, 0, 1);
            __builtin_prefetch(B + (size_t)(n0 + t) * K + kc + BK, 0, 1);
        }
        compute64(As[buf], Bs[buf], msub, nsub, lg, lr, acc);   // WMMAs overlap copies
        wait_async0();
        __syncthreads();
        buf ^= 1;
    }
    compute64(As[buf], Bs[buf], msub, nsub, lg, lr, acc);

    // D layout: VGPR rr -> M=rr (lanes 0-15), M=rr+8 (lanes 16-31); N = lane&15
    #pragma unroll
    for (int mt = 0; mt < 2; ++mt)
        #pragma unroll
        for (int nt = 0; nt < 4; ++nt)
            #pragma unroll
            for (int rr = 0; rr < 8; ++rr) {
                int m = m0 + msub + mt * 16 + rr + lg * 8;
                int n = n0 + nsub + nt * 16 + lr;
                C[(size_t)m * N + n] = acc[mt][nt][rr];
            }
}

// ---------------------------------------------------------------------------
// RoPE (interleaved pairs) in f32, emit bf16.  Q: [s][32][64]
// ---------------------------------------------------------------------------
__global__ void llama_rope_q(const float* __restrict__ Qf,
                             unsigned short* __restrict__ Qb) {
    int p = blockIdx.x * blockDim.x + threadIdx.x;    // pair index
    if (p >= SEQ * NHEADS * (HDIM / 2)) return;
    int d2 = p & 31;
    int h  = (p >> 5) & 31;
    int s  = p >> 10;
    float inv = __powf(10000.0f, -(float)d2 / 32.0f);
    float ang = (float)s * inv;
    float c = __cosf(ang), sn = __sinf(ang);
    size_t base = (size_t)s * HID + h * HDIM + 2 * d2;
    float x0 = Qf[base], x1 = Qf[base + 1];
    Qb[base]     = f2bf(x0 * c - x1 * sn);
    Qb[base + 1] = f2bf(x1 * c + x0 * sn);
}

// K: [s][8][64]
__global__ void llama_rope_k(const float* __restrict__ Kf,
                             unsigned short* __restrict__ Kb) {
    int p = blockIdx.x * blockDim.x + threadIdx.x;
    if (p >= SEQ * NKV * (HDIM / 2)) return;
    int d2 = p & 31;
    int h  = (p >> 5) & 7;
    int s  = p >> 8;
    float inv = __powf(10000.0f, -(float)d2 / 32.0f);
    float ang = (float)s * inv;
    float c = __cosf(ang), sn = __sinf(ang);
    size_t base = (size_t)s * KVDIM + h * HDIM + 2 * d2;
    float x0 = Kf[base], x1 = Kf[base + 1];
    Kb[base]     = f2bf(x0 * c - x1 * sn);
    Kb[base + 1] = f2bf(x1 * c + x0 * sn);
}

// V: [s][8][64] f32  ->  Vt bf16 [8][64][s]  (so P*V B-fragments are contiguous)
__global__ void llama_cvt_v(const float* __restrict__ Vf,
                            unsigned short* __restrict__ Vt) {
    int i = blockIdx.x * blockDim.x + threadIdx.x;
    if (i >= SEQ * KVDIM) return;
    int s = i >> 9;
    int r = i & 511;          // h*64 + d
    Vt[((size_t)r << 11) + s] = f2bf(Vf[i]);
}

// ---------------------------------------------------------------------------
// Flash attention, causal, GQA.  1 wave per (16-query block, head).
// Register double-buffered: next chunk's K/V fragments are issued right after
// the score WMMAs so their latency hides under the softmax VALU work.
// ---------------------------------------------------------------------------
__device__ __forceinline__ void load_kfr(const unsigned short* Kh, int kc,
                                         int lg, int lr, v16bf kf[2][2]) {
    #pragma unroll
    for (int t = 0; t < 2; ++t)
        #pragma unroll
        for (int st = 0; st < 2; ++st) {
            BF16Frag b;   // 32x16: contraction over head-dim chunk, N = key
            const unsigned short* p =
                Kh + (size_t)(kc + t * 16 + lr) * KVDIM + st * 32 + lg * 16;
            b.q[0] = ((const u32x4*)p)[0];
            b.q[1] = ((const u32x4*)p)[1];
            kf[t][st] = b.v;
        }
}

__device__ __forceinline__ void load_vfr(const unsigned short* Vh, int kc,
                                         int lg, int lr, v16bf vf[4]) {
    #pragma unroll
    for (int t = 0; t < 4; ++t) {
        BF16Frag b;   // 32x16: contraction over 32 keys, N = head-dim col
        const unsigned short* p = Vh + ((size_t)(t * 16 + lr) << 11) + kc + lg * 16;
        b.q[0] = ((const u32x4*)p)[0];
        b.q[1] = ((const u32x4*)p)[1];
        vf[t] = b.v;
    }
}

__device__ __forceinline__ void attn_chunk(
    int k0, bool more, int q0, int lg, int lr, int l,
    const v16bf qa[2], const v16bf kcur[2][2], const v16bf vcur[4],
    v16bf knxt[2][2], v16bf vnxt[4],
    const unsigned short* Kh, const unsigned short* Vh,
    float& m_run, float& l_run, v8f o[4],
    float (*sp)[32], float* bcast, float scale) {

    // ---- scores: two 16x16 tiles (keys k0..k0+15, k0+16..k0+31) ----
    #pragma unroll
    for (int t = 0; t < 2; ++t) {
        v8f s = zero8();
        #pragma unroll
        for (int st = 0; st < 2; ++st)
            s = __builtin_amdgcn_wmma_f32_16x16x32_bf16(
                false, qa[st], false, kcur[t][st], (short)0, s, false, false);
        const int key = k0 + t * 16 + lr;
        #pragma unroll
        for (int r = 0; r < 8; ++r) {
            int m = r + lg * 8;
            sp[m][t * 16 + lr] =
                (key <= q0 + m) ? s[r] * scale : -__builtin_inff();
        }
    }

    // ---- issue next chunk's K/V loads; latency hides under softmax ----
    if (more) {
        load_kfr(Kh, k0 + 32, lg, lr, knxt);
        load_vfr(Vh, k0 + 32, lg, lr, vnxt);
    }
    __syncthreads();

    // ---- online softmax stats: lane handles row lr, half lg ----
    float cmax = -__builtin_inff();
    #pragma unroll
    for (int j = 0; j < 16; ++j) cmax = fmaxf(cmax, sp[lr][lg * 16 + j]);
    cmax = fmaxf(cmax, __shfl_xor(cmax, 16, 32));
    const float mnew  = fmaxf(m_run, cmax);          // finite after chunk 0
    const float alpha = __expf(m_run - mnew);        // 0 on first chunk

    float csum = 0.0f;
    #pragma unroll
    for (int j = 0; j < 16; ++j) {
        float pv = __expf(sp[lr][lg * 16 + j] - mnew);
        sp[lr][lg * 16 + j] = pv;
        csum += pv;
    }
    csum += __shfl_xor(csum, 16, 32);
    l_run = l_run * alpha + csum;
    m_run = mnew;

    if (l < 16) bcast[lr] = alpha;
    __syncthreads();
    float ar[8];
    #pragma unroll
    for (int r = 0; r < 8; ++r) ar[r] = bcast[r + lg * 8];
    #pragma unroll
    for (int t = 0; t < 4; ++t)
        #pragma unroll
        for (int r = 0; r < 8; ++r) o[t][r] *= ar[r];

    // ---- P (16x32) fragment from LDS ----
    BF16Frag pa;
    #pragma unroll
    for (int j = 0; j < 8; ++j) {
        pa.h[j]     = f2bf(sp[lr][lg * 16 + j]);
        pa.h[8 + j] = f2bf(sp[lr][lg * 16 + 8 + j]);
    }

    // ---- O += P @ V : 4 tiles over head dim ----
    #pragma unroll
    for (int t = 0; t < 4; ++t)
        o[t] = __builtin_amdgcn_wmma_f32_16x16x32_bf16(
            false, pa.v, false, vcur[t], (short)0, o[t], false, false);
    __syncthreads();   // before next chunk overwrites sp
}

__global__ __launch_bounds__(32)
void llama_attn(const unsigned short* __restrict__ Qb,
                const unsigned short* __restrict__ Kb,
                const unsigned short* __restrict__ Vt,
                unsigned short* __restrict__ Ob) {
    __shared__ float sp[16][32];     // score / prob chunk
    __shared__ float bcast[16];      // per-row broadcast (alpha / inv-sum)

    const int qb  = blockIdx.x;
    const int h   = blockIdx.y;
    const int hkv = h >> 2;          // GQA: 4 query heads per KV head
    const int q0  = qb * 16;
    const int l   = threadIdx.x;
    const int lg  = l >> 4;
    const int lr  = l & 15;
    const float scale = 0.125f;      // 1/sqrt(64)

    // Q fragments: two k-steps over head dim (d 0..31 and 32..63)
    v16bf qa[2];
    const unsigned short* qrow = Qb + ((size_t)(q0 + lr) * NHEADS + h) * HDIM;
    #pragma unroll
    for (int st = 0; st < 2; ++st) {
        BF16Frag a;
        a.q[0] = *(const u32x4*)(qrow + st * 32 + lg * 16);
        a.q[1] = *(const u32x4*)(qrow + st * 32 + 8 + lg * 16);
        qa[st] = a.v;
    }

    v8f o[4];
    #pragma unroll
    for (int t = 0; t < 4; ++t) o[t] = zero8();
    float m_run = -__builtin_inff();
    float l_run = 0.0f;

    const int nk = q0 + 16;          // causal horizon for this query block
    const unsigned short* Kh = Kb + hkv * HDIM;
    const unsigned short* Vh = Vt + ((size_t)(hkv * HDIM) << 11);

    v16bf ka[2][2], va[4], kb2[2][2], vb2[4];
    load_kfr(Kh, 0, lg, lr, ka);
    load_vfr(Vh, 0, lg, lr, va);

    int k0 = 0;
    while (true) {
        bool more = (k0 + 32) < nk;
        attn_chunk(k0, more, q0, lg, lr, l, qa, ka, va, kb2, vb2,
                   Kh, Vh, m_run, l_run, o, sp, bcast, scale);
        k0 += 32;
        if (!more) break;
        bool more2 = (k0 + 32) < nk;
        attn_chunk(k0, more2, q0, lg, lr, l, qa, kb2, vb2, ka, va,
                   Kh, Vh, m_run, l_run, o, sp, bcast, scale);
        k0 += 32;
        if (!more2) break;
    }

    // ---- normalize rows and store bf16 [s][h*64+d] ----
    if (l < 16) bcast[lr] = 1.0f / l_run;
    __syncthreads();
    float ir[8];
    #pragma unroll
    for (int r = 0; r < 8; ++r) ir[r] = bcast[r + lg * 8];
    #pragma unroll
    for (int t = 0; t < 4; ++t) {
        #pragma unroll
        for (int r = 0; r < 8; ++r) {
            int m = r + lg * 8;
            int d = t * 16 + lr;
            Ob[(size_t)(q0 + m) * HID + h * HDIM + d] = f2bf(o[t][r] * ir[r]);
        }
    }
}

// ---------------------------------------------------------------------------
// launch
// ---------------------------------------------------------------------------
extern "C" void kernel_launch(void* const* d_in, const int* in_sizes, int n_in,
                              void* d_out, int out_size, void* d_ws, size_t ws_size,
                              hipStream_t stream) {
    const float* X  = (const float*)d_in[0];  // [2048, 2048]
    const float* Wq = (const float*)d_in[1];  // [2048, 2048]
    const float* Wk = (const float*)d_in[2];  // [512, 2048]
    const float* Wv = (const float*)d_in[3];  // [512, 2048]
    const float* Wo = (const float*)d_in[4];  // [2048, 2048]
    float* Out = (float*)d_out;               // [2048, 2048]

    // workspace carve-out (all regions 256B aligned)
    char* base = (char*)d_ws;
    size_t off = 0;
    auto carve = [&](size_t bytes) {
        void* p = base + off;
        off += (bytes + 255) & ~(size_t)255;
        return p;
    };
    unsigned short* Xb  = (unsigned short*)carve((size_t)SEQ * HID * 2);
    unsigned short* Wqb = (unsigned short*)carve((size_t)HID * HID * 2);
    unsigned short* Wkb = (unsigned short*)carve((size_t)KVDIM * HID * 2);
    unsigned short* Wvb = (unsigned short*)carve((size_t)KVDIM * HID * 2);
    unsigned short* Wob = (unsigned short*)carve((size_t)HID * HID * 2);
    float*          Qf  = (float*)carve((size_t)SEQ * HID * 4);
    float*          Kf  = (float*)carve((size_t)SEQ * KVDIM * 4);
    float*          Vf  = (float*)carve((size_t)SEQ * KVDIM * 4);
    unsigned short* Qb  = (unsigned short*)carve((size_t)SEQ * HID * 2);
    unsigned short* Kb  = (unsigned short*)carve((size_t)SEQ * KVDIM * 2);
    unsigned short* Vt  = (unsigned short*)carve((size_t)SEQ * KVDIM * 2);
    unsigned short* Ab  = (unsigned short*)carve((size_t)SEQ * HID * 2);

    const int T = 256;
    // 1) convert inputs to bf16
    llama_cvt_bf16<<<(SEQ * HID + T - 1) / T, T, 0, stream>>>(X, Xb, SEQ * HID);
    llama_cvt_bf16<<<(HID * HID + T - 1) / T, T, 0, stream>>>(Wq, Wqb, HID * HID);
    llama_cvt_bf16<<<(KVDIM * HID + T - 1) / T, T, 0, stream>>>(Wk, Wkb, KVDIM * HID);
    llama_cvt_bf16<<<(KVDIM * HID + T - 1) / T, T, 0, stream>>>(Wv, Wvb, KVDIM * HID);
    llama_cvt_bf16<<<(HID * HID + T - 1) / T, T, 0, stream>>>(Wo, Wob, HID * HID);

    // 2) QKV projections (C = A @ W^T), f32 accum
    llama_gemm_bf16<<<dim3(HID / BN, SEQ / BM), 128, 0, stream>>>(Xb, Wqb, Qf, SEQ, HID, HID);
    llama_gemm_bf16<<<dim3(KVDIM / BN, SEQ / BM), 128, 0, stream>>>(Xb, Wkb, Kf, SEQ, KVDIM, HID);
    llama_gemm_bf16<<<dim3(KVDIM / BN, SEQ / BM), 128, 0, stream>>>(Xb, Wvb, Vf, SEQ, KVDIM, HID);

    // 3) RoPE + bf16 conversion; V transpose to [h][d][s]
    llama_rope_q<<<(SEQ * NHEADS * (HDIM / 2) + T - 1) / T, T, 0, stream>>>(Qf, Qb);
    llama_rope_k<<<(SEQ * NKV * (HDIM / 2) + T - 1) / T, T, 0, stream>>>(Kf, Kb);
    llama_cvt_v<<<(SEQ * KVDIM + T - 1) / T, T, 0, stream>>>(Vf, Vt);

    // 4) flash attention (causal, GQA)
    llama_attn<<<dim3(SEQ / 16, NHEADS), 32, 0, stream>>>(Qb, Kb, Vt, Ab);

    // 5) output projection -> f32
    llama_gemm_bf16<<<dim3(HID / BN, SEQ / BM), 128, 0, stream>>>(Ab, Wob, Out, SEQ, HID, HID);
}
